// MultiHeadAttention_56349970923837
// MI455X (gfx1250) — matrix operand order repaired and checked by reading, not compiled, not run on hardware
//
#include <hip/hip_runtime.h>
#include <hip/hip_bf16.h>
#include <stdint.h>

#define B_   2
#define T_   4096
#define H_   8
#define DH_  64
#define C_   512
#define N3_  1536
#define M_   (B_ * T_)

typedef __attribute__((ext_vector_type(16))) __bf16 v16bf;
typedef __attribute__((ext_vector_type(8)))  float  v8f;
typedef __attribute__((ext_vector_type(4)))  int    v4i;
typedef unsigned short u16;

// ---------------- async global->LDS (CDNA5) with safe fallback ----------------
#ifdef __has_builtin
#if __has_builtin(__builtin_amdgcn_global_load_async_to_lds_b128)
#define USE_ASYNC_LDS 1
#endif
#endif
#ifndef USE_ASYNC_LDS
#define USE_ASYNC_LDS 0
#endif

#if USE_ASYNC_LDS
typedef __attribute__((address_space(1))) v4i* gv4i_p;   // global int4*
typedef __attribute__((address_space(3))) v4i* lv4i_p;   // LDS int4*
__device__ __forceinline__ void async_copy16(void* lds, const void* g) {
  void* gg = const_cast<void*>(g);
  __builtin_amdgcn_global_load_async_to_lds_b128((gv4i_p)gg, (lv4i_p)lds, 0, 0);
}
#define ASYNC_WAIT() asm volatile("s_wait_asynccnt 0" ::: "memory")
#else
__device__ __forceinline__ void async_copy16(void* lds, const void* g) {
  *(uint4*)lds = *(const uint4*)g;
}
#define ASYNC_WAIT()
#endif

// ---------------- helpers ----------------
__device__ __forceinline__ u16 f2bf(float x) {
  unsigned u = __builtin_bit_cast(unsigned, x);
  u += 0x7FFFu + ((u >> 16) & 1u);
  return (u16)(u >> 16);
}

union FragBF { v16bf bf; uint4 q[2]; };

__device__ __forceinline__ v8f vzero8() {
  v8f v;
#pragma unroll
  for (int i = 0; i < 8; ++i) v[i] = 0.0f;
  return v;
}

// A fragment (16x32 bf16): lane l<16 -> row l, K {0..7,16..23}; lane>=16 -> row l-16, K {8..15,24..31}
__device__ __forceinline__ FragBF load_a_frag(const u16* base, int row, int ldw, int lane) {
  const u16* rp = base + row * ldw + ((lane & 16) ? 8 : 0);
  FragBF f;
  f.q[0] = *(const uint4*)(rp);
  f.q[1] = *(const uint4*)(rp + 16);
  return f;
}

// B fragment (32x16 bf16) from N-major (transposed) tile Bt[n][k]:
// lane l<16 -> column l, K {0..7,16..23}; lane>=16 -> column l-16, K {8..15,24..31}
__device__ __forceinline__ FragBF load_b_frag(const u16* base, int col, int ldw, int lane, int kbase) {
  const u16* rp = base + col * ldw + kbase + ((lane & 16) ? 8 : 0);
  FragBF f;
  f.q[0] = *(const uint4*)(rp);
  f.q[1] = *(const uint4*)(rp + 16);
  return f;
}

__device__ __forceinline__ v8f wmma_bf16(const FragBF& a, const FragBF& b, v8f c) {
  return __builtin_amdgcn_wmma_f32_16x16x32_bf16(false, a.bf, false, b.bf, (short)0, c, false, false);
}

// ---------------- kernel 1: fp32 -> bf16 conversion + weight transposes ----------------
__global__ void mha_prep(const float* __restrict__ x, const float* __restrict__ wqkv,
                         const float* __restrict__ wout, u16* __restrict__ xbf,
                         u16* __restrict__ wqkvT, u16* __restrict__ woutT) {
  const int NX = M_ * C_;       // 4194304
  const int NQ = C_ * N3_;      // 786432
  const int NO = C_ * C_;       // 262144
  const int NT = NX + NQ + NO;
  for (int i = blockIdx.x * blockDim.x + threadIdx.x; i < NT; i += gridDim.x * blockDim.x) {
    if (i < NX) {
      xbf[i] = f2bf(x[i]);
    } else if (i < NX + NQ) {
      int j = i - NX;
      int k = j / N3_, n = j - k * N3_;
      wqkvT[(size_t)n * C_ + k] = f2bf(wqkv[j]);
    } else {
      int j = i - NX - NQ;
      int k = j >> 9, n = j & (C_ - 1);
      woutT[(size_t)n * C_ + k] = f2bf(wout[j]);
    }
  }
}

// ---------------- kernel 2: QKV projection GEMM (M=8192, N=1536, K=512) ----------------
// 128x64 block tile, 8 waves, each wave a 16x64 strip. Q scaled by 1/sqrt(Dh), V stored transposed.
__global__ __launch_bounds__(256) void mha_qkv_gemm(
    const u16* __restrict__ xbf, const u16* __restrict__ wqkvT, const float* __restrict__ bqkv,
    u16* __restrict__ Q, u16* __restrict__ K, u16* __restrict__ Vt) {
  __shared__ u16 Al[128][40];
  __shared__ u16 Bl[64][40];
  const int tid = threadIdx.x, lane = tid & 31, wave = tid >> 5;
  const int m0 = blockIdx.x * 128, n0 = blockIdx.y * 64;
  const int lnlo = lane & 15, hi8 = (lane >> 4) * 8;

  v8f acc[4];
#pragma unroll
  for (int i = 0; i < 4; ++i) acc[i] = vzero8();

  for (int k0 = 0; k0 < C_; k0 += 32) {
    {
      int row = tid >> 1, cc = (tid & 1) * 16;
      const u16* g = xbf + (size_t)(m0 + row) * C_ + k0 + cc;
      async_copy16(&Al[row][cc], g);
      async_copy16(&Al[row][cc + 8], g + 8);
    }
    {
      int row = tid >> 2, cc = (tid & 3) * 8;
      const u16* g = wqkvT + (size_t)(n0 + row) * C_ + k0 + cc;
      async_copy16(&Bl[row][cc], g);
    }
    ASYNC_WAIT();
    __syncthreads();

    FragBF a = load_a_frag(&Al[0][0], wave * 16 + lnlo, 40, lane);
#pragma unroll
    for (int nt = 0; nt < 4; ++nt) {
      FragBF b = load_b_frag(&Bl[0][0], nt * 16 + lnlo, 40, lane, 0);
      acc[nt] = wmma_bf16(a, b, acc[nt]);
    }
    __syncthreads();
  }

#pragma unroll
  for (int nt = 0; nt < 4; ++nt) {
    int j = n0 + nt * 16 + lnlo;           // 0..1535
    float bias = bqkv[j];
    int which = j >> 9;                    // 0=Q, 1=K, 2=V
    int jj = j & 511, h = jj >> 6, d = jj & 63;
#pragma unroll
    for (int r = 0; r < 8; ++r) {
      int row = m0 + wave * 16 + r + hi8;  // global token row in [0, 8192)
      int b_ = row >> 12, tt = row & (T_ - 1);
      int bh = b_ * H_ + h;
      float v = acc[nt][r] + bias;
      if (which == 0)      Q[((size_t)bh * T_ + tt) * DH_ + d] = f2bf(v * 0.125f);
      else if (which == 1) K[((size_t)bh * T_ + tt) * DH_ + d] = f2bf(v);
      else                 Vt[((size_t)bh * DH_ + d) * T_ + tt] = f2bf(v);
    }
  }
}

// ---------------- kernel 3: causal flash attention ----------------
// grid = (T/64, B*H), 128 threads = 4 waves; wave w owns query rows [q0+16w, q0+16w+16)
__global__ __launch_bounds__(128) void mha_attn(
    const u16* __restrict__ Q, const u16* __restrict__ K, const u16* __restrict__ Vt,
    u16* __restrict__ y) {
  __shared__ u16 Kl[32][72];      // [key][dh]     == Bt for S = Q*K^T
  __shared__ u16 Vl[64][40];      // [dh][key]     == Bt for O = P*V
  __shared__ u16 Pl[4][16][40];   // per-wave P reshape buffer (C-layout -> A-layout)

  const int tid = threadIdx.x, lane = tid & 31, wave = tid >> 5;
  const int bh = blockIdx.y, b_ = bh >> 3, h = bh & 7;
  const int q0 = blockIdx.x * 64, qw0 = q0 + wave * 16;
  const int lnlo = lane & 15, hi8 = (lane >> 4) * 8;

  // Q A-fragments for Dh=64 (two K-chunks of 32), loaded straight from global
  const u16* Qrow = Q + ((size_t)bh * T_ + qw0 + lnlo) * DH_;
  FragBF aq[2];
#pragma unroll
  for (int kk = 0; kk < 2; ++kk) {
    int klo = (lane & 16) ? 8 : 0;
    aq[kk].q[0] = *(const uint4*)(Qrow + kk * 32 + klo);
    aq[kk].q[1] = *(const uint4*)(Qrow + kk * 32 + 16 + klo);
  }

  v8f o[4];
  float mrow[8], lrow[8];
#pragma unroll
  for (int nt = 0; nt < 4; ++nt) o[nt] = vzero8();
#pragma unroll
  for (int r = 0; r < 8; ++r) { mrow[r] = -1e30f; lrow[r] = 0.0f; }

  const int nkb = blockIdx.x * 2 + 2;     // key blocks of 32, causal upper bound for this WG
  for (int kb = 0; kb < nkb; ++kb) {
    const int k0 = kb * 32;
    {   // stage K tile 32x64
      int row = tid >> 2, cc = (tid & 3) * 16;
      const u16* g = K + ((size_t)bh * T_ + k0 + row) * DH_ + cc;
      async_copy16(&Kl[row][cc], g);
      async_copy16(&Kl[row][cc + 8], g + 8);
    }
    {   // stage V^T tile 64x32
      int row = tid >> 1, cc = (tid & 1) * 16;
      const u16* g = Vt + ((size_t)bh * DH_ + row) * T_ + k0 + cc;
      async_copy16(&Vl[row][cc], g);
      async_copy16(&Vl[row][cc + 8], g + 8);
    }
    ASYNC_WAIT();
    __syncthreads();

    if (k0 < qw0 + 16) {                  // this wave has unmasked work in this key block
      v8f sc[2];
      sc[0] = vzero8(); sc[1] = vzero8();
#pragma unroll
      for (int kk = 0; kk < 2; ++kk)
#pragma unroll
        for (int nt = 0; nt < 2; ++nt) {
          FragBF bk = load_b_frag(&Kl[0][0], nt * 16 + lnlo, 72, lane, kk * 32);
          sc[nt] = wmma_bf16(aq[kk], bk, sc[nt]);
        }

      const bool need_mask = (k0 + 31 > qw0);
      float fsc[8];
#pragma unroll
      for (int r = 0; r < 8; ++r) {
        float s0 = sc[0][r], s1 = sc[1][r];
        if (need_mask) {
          int qrow = qw0 + r + hi8;
          if (k0 + lnlo > qrow)      s0 = -1e30f;
          if (k0 + 16 + lnlo > qrow) s1 = -1e30f;
        }
        // row max / sum across the 16 lanes holding this row's columns
        float v = fmaxf(s0, s1);
#pragma unroll
        for (int m = 8; m >= 1; m >>= 1) v = fmaxf(v, __shfl_xor(v, m, 32));
        float nm = fmaxf(mrow[r], v);
        float p0 = __expf(s0 - nm), p1 = __expf(s1 - nm);
        float rs = p0 + p1;
#pragma unroll
        for (int m = 8; m >= 1; m >>= 1) rs += __shfl_xor(rs, m, 32);
        float scal = __expf(mrow[r] - nm);
        lrow[r] = lrow[r] * scal + rs;
        mrow[r] = nm;
        fsc[r] = scal;
        Pl[wave][r + hi8][lnlo]      = f2bf(p0);
        Pl[wave][r + hi8][lnlo + 16] = f2bf(p1);
      }
#pragma unroll
      for (int nt = 0; nt < 4; ++nt)
#pragma unroll
        for (int r = 0; r < 8; ++r) o[nt][r] *= fsc[r];

      // wave-private LDS round trip: writes above -> A-fragment reads below
      asm volatile("s_wait_dscnt 0" ::: "memory");
      FragBF ap = load_a_frag(&Pl[wave][0][0], lnlo, 40, lane);
#pragma unroll
      for (int nt = 0; nt < 4; ++nt) {
        FragBF bv = load_b_frag(&Vl[0][0], nt * 16 + lnlo, 40, lane, 0);
        o[nt] = wmma_bf16(ap, bv, o[nt]);
      }
    }
    __syncthreads();
  }

  float inv[8];
#pragma unroll
  for (int r = 0; r < 8; ++r) inv[r] = 1.0f / lrow[r];
#pragma unroll
  for (int nt = 0; nt < 4; ++nt)
#pragma unroll
    for (int r = 0; r < 8; ++r) {
      int qrow = qw0 + r + hi8;
      int dh = nt * 16 + lnlo;
      y[((size_t)(b_ * T_ + qrow)) * C_ + h * DH_ + dh] = f2bf(o[nt][r] * inv[r]);
    }
}

// ---------------- kernel 4: output projection GEMM (M=8192, N=512, K=512), fp32 out ----------------
__global__ __launch_bounds__(256) void mha_out_gemm(
    const u16* __restrict__ ybf, const u16* __restrict__ woutT, const float* __restrict__ bout,
    float* __restrict__ out) {
  __shared__ u16 Al[128][40];
  __shared__ u16 Bl[64][40];
  const int tid = threadIdx.x, lane = tid & 31, wave = tid >> 5;
  const int m0 = blockIdx.x * 128, n0 = blockIdx.y * 64;
  const int lnlo = lane & 15, hi8 = (lane >> 4) * 8;

  v8f acc[4];
#pragma unroll
  for (int i = 0; i < 4; ++i) acc[i] = vzero8();

  for (int k0 = 0; k0 < C_; k0 += 32) {
    {
      int row = tid >> 1, cc = (tid & 1) * 16;
      const u16* g = ybf + (size_t)(m0 + row) * C_ + k0 + cc;
      async_copy16(&Al[row][cc], g);
      async_copy16(&Al[row][cc + 8], g + 8);
    }
    {
      int row = tid >> 2, cc = (tid & 3) * 8;
      const u16* g = woutT + (size_t)(n0 + row) * C_ + k0 + cc;
      async_copy16(&Bl[row][cc], g);
    }
    ASYNC_WAIT();
    __syncthreads();

    FragBF a = load_a_frag(&Al[0][0], wave * 16 + lnlo, 40, lane);
#pragma unroll
    for (int nt = 0; nt < 4; ++nt) {
      FragBF b = load_b_frag(&Bl[0][0], nt * 16 + lnlo, 40, lane, 0);
      acc[nt] = wmma_bf16(a, b, acc[nt]);
    }
    __syncthreads();
  }

#pragma unroll
  for (int nt = 0; nt < 4; ++nt) {
    int j = n0 + nt * 16 + lnlo;
    float bias = bout[j];
#pragma unroll
    for (int r = 0; r < 8; ++r) {
      int row = m0 + wave * 16 + r + hi8;
      out[(size_t)row * C_ + j] = acc[nt][r] + bias;
    }
  }
}

// ---------------- host launcher ----------------
extern "C" void kernel_launch(void* const* d_in, const int* in_sizes, int n_in,
                              void* d_out, int out_size, void* d_ws, size_t ws_size,
                              hipStream_t stream) {
  (void)in_sizes; (void)n_in; (void)out_size; (void)ws_size;
  const float* x    = (const float*)d_in[0];
  const float* wqkv = (const float*)d_in[1];
  const float* bqkv = (const float*)d_in[2];
  const float* wout = (const float*)d_in[3];
  const float* bout = (const float*)d_in[4];
  float* out = (float*)d_out;

  char* ws = (char*)d_ws;
  // workspace layout (bytes), all 256-aligned; x_bf region is reused as y after QKV GEMM
  u16* xbf_y = (u16*)(ws + 0);                    //  8,388,608  (x in bf16, later y in bf16)
  u16* wqkvT = (u16*)(ws + 8388608);              //  1,572,864
  u16* woutT = (u16*)(ws + 9961472);              //    524,288
  u16* Qb    = (u16*)(ws + 10485760);             //  8,388,608  [B,H,T,Dh] bf16, pre-scaled
  u16* Kb    = (u16*)(ws + 18874368);             //  8,388,608  [B,H,T,Dh] bf16
  u16* Vtb   = (u16*)(ws + 27262976);             //  8,388,608  [B,H,Dh,T] bf16 (transposed)
  // total: 35,651,584 bytes

  mha_prep<<<4096, 256, 0, stream>>>(x, wqkv, wout, xbf_y, wqkvT, woutT);
  mha_qkv_gemm<<<dim3(M_ / 128, N3_ / 64), 256, 0, stream>>>(xbf_y, wqkvT, bqkv, Qb, Kb, Vtb);
  mha_attn<<<dim3(T_ / 64, B_ * H_), 128, 0, stream>>>(Qb, Kb, Vtb, xbf_y);
  mha_out_gemm<<<dim3(M_ / 128, C_ / 64), 256, 0, stream>>>(xbf_y, woutT, bout, out);
}
